// GRU_91216515432821
// MI455X (gfx1250) — matrix operand order repaired and checked by reading, not compiled, not run on hardware
//
#include <hip/hip_runtime.h>
#include <hip/hip_bf16.h>

// GRU: X[S,B] int tokens, h[B,DL] f32, emb[V,DI] f32, W_z/W_r/W_h [DL+DI, DL] f32
// Output H [S,B,DL] f32.
//
//   prep:   convert emb + weights to bf16; weights stored transposed [N][K].
//   uproj:  U_g = emb[X] @ W_g[DL:,:]  for all 3 gates per wave (shared A frag),
//           stored in WMMA C-fragment-swizzled layout (coalesced 16B stores).
//   recur:  persistent kernel (32 blocks), 512 steps, 2 grid barriers/step.
//           phase1: z,r fused per wave (shared A frag, dual WMMA chains).
//           phase2: cand + state update, dual accumulators over K.

typedef __attribute__((ext_vector_type(16))) __bf16 v16bf;
typedef __attribute__((ext_vector_type(8)))  __bf16 v8bf;
typedef __attribute__((ext_vector_type(8)))  float  v8f;

constexpr int kSEQ = 512;
constexpr int kB   = 64;
constexpr int kDI  = 512;
constexpr int kDL  = 1024;
constexpr int kV   = 50257;
constexpr int kNT  = kDL / 16;          // 64 column tiles
constexpr int kMTg = (kSEQ * kB) / 16;  // 2048 global row tiles for U

// ---------------- WMMA fragment loaders (bf16, 16x16x32) ------------------
// A (16x32, row-major src, leading dim ld): lane l (row m=l&15):
//   elems 0..7 = K k0..k0+7, elems 8..15 = K k0+16..k0+23, k0=(l<16)?0:8.
__device__ __forceinline__ v16bf load_a_frag(const __bf16* tile, int ld, int lane) {
  const int m  = lane & 15;
  const int k0 = (lane & 16) ? 8 : 0;
  const __bf16* p = tile + (size_t)m * ld + k0;
  v8bf lo = *(const v8bf*)(p);
  v8bf hi = *(const v8bf*)(p + 16);
  v16bf r;
#pragma unroll
  for (int i = 0; i < 8; ++i) { r[i] = lo[i]; r[i + 8] = hi[i]; }
  return r;
}

// B (32x16) from transposed weights Wt[N][K]: lane n: elems 0..7 =
//   K k0..k0+7, elems 8..15 = K k0+16..k0+23, k0=(l<16)?0:8.  Two 16B loads.
__device__ __forceinline__ v16bf load_b_frag(const __bf16* tile, int ld, int lane) {
  const int n  = lane & 15;
  const int k0 = (lane & 16) ? 8 : 0;
  const __bf16* p = tile + (size_t)n * ld + k0;
  v8bf lo = *(const v8bf*)(p);
  v8bf hi = *(const v8bf*)(p + 16);
  v16bf r;
#pragma unroll
  for (int i = 0; i < 8; ++i) { r[i] = lo[i]; r[i + 8] = hi[i]; }
  return r;
}

// Gathered A rows via token ids (emb bf16 [V][DI] row-major).
__device__ __forceinline__ v16bf load_a_gather(const __bf16* embbf, const int* X,
                                               int row0, int kbase, int lane) {
  const int m  = lane & 15;
  const int k0 = (lane & 16) ? 8 : 0;
  const int tok = X[row0 + m];
  const __bf16* p = embbf + (size_t)tok * kDI + kbase + k0;
  v8bf lo = *(const v8bf*)(p);
  v8bf hi = *(const v8bf*)(p + 16);
  v16bf r;
#pragma unroll
  for (int i = 0; i < 8; ++i) { r[i] = lo[i]; r[i + 8] = hi[i]; }
  return r;
}

__device__ __forceinline__ v8f wmma_bf16(v16bf a, v16bf b, v8f c) {
  return __builtin_amdgcn_wmma_f32_16x16x32_bf16(false, a, false, b, (short)0, c,
                                                 false, false);
}

// C/D fragment swizzle: tile-major, then lane*8+e. Contiguous per lane.
__device__ __forceinline__ int swz_off(int m, int n) {  // canonical -> swizzled
  int mt = m >> 4, nt = n >> 4, ml = m & 15, nl = n & 15;
  int lane = nl + ((ml & 8) ? 16 : 0);
  int e = ml & 7;
  return (((mt << 6) + nt) << 8) + (lane << 3) + e;     // 64 col tiles (DL)
}

// ---------------- software grid barrier (device-scope atomics) -------------
__device__ __forceinline__ void grid_barrier(unsigned* c, unsigned nblk) {
  __syncthreads();
  if (threadIdx.x == 0) {
    __threadfence();  // release
    __hip_atomic_fetch_add(c, 1u, __ATOMIC_RELEASE, __HIP_MEMORY_SCOPE_AGENT);
    while (__hip_atomic_load(c, __ATOMIC_ACQUIRE, __HIP_MEMORY_SCOPE_AGENT) < nblk) {
      __builtin_amdgcn_s_sleep(1);
    }
  }
  __syncthreads();
  __threadfence();  // acquire
}

// ---------------- prep kernels ---------------------------------------------
__global__ void k_emb2bf(const float* __restrict__ emb, __bf16* __restrict__ embbf,
                         long n) {
  long i = (long)blockIdx.x * blockDim.x + threadIdx.x;
  if (i < n) embbf[i] = (__bf16)emb[i];
}

__global__ void k_wx(const float* __restrict__ Wz, const float* __restrict__ Wr,
                     const float* __restrict__ Wh, __bf16* __restrict__ Wxt) {
  long i = (long)blockIdx.x * blockDim.x + threadIdx.x;
  const long per = (long)kDL * kDI;
  if (i < 3 * per) {
    int g = (int)(i / per);
    long r = i % per;
    int n = (int)(r / kDI), k = (int)(r % kDI);
    const float* W = (g == 0) ? Wz : ((g == 1) ? Wr : Wh);
    Wxt[i] = (__bf16)W[(size_t)(kDL + k) * kDL + n];
  }
}

__global__ void k_wh(const float* __restrict__ Wz, const float* __restrict__ Wr,
                     const float* __restrict__ Wh, __bf16* __restrict__ Wht) {
  long i = (long)blockIdx.x * blockDim.x + threadIdx.x;
  const long per = (long)kDL * kDL;
  if (i < 3 * per) {
    int g = (int)(i / per);
    long r = i % per;
    int n = (int)(r / kDL), k = (int)(r % kDL);
    const float* W = (g == 0) ? Wz : ((g == 1) ? Wr : Wh);
    Wht[i] = (__bf16)W[(size_t)k * kDL + n];
  }
}

// init: zero barrier counters, seed swizzled f32 h + canonical bf16 h.
__global__ void k_init(const float* __restrict__ h, float* __restrict__ hsw0,
                       __bf16* __restrict__ hbf0, unsigned* __restrict__ cnt) {
  int i = blockIdx.x * blockDim.x + threadIdx.x;
  if (i < 2 * kSEQ) cnt[i] = 0u;
  if (i < kB * kDL) {
    float v = h[i];
    hsw0[swz_off(i / kDL, i % kDL)] = v;
    hbf0[i] = (__bf16)v;
  }
}

// ---------------- input-projection GEMM (parallel over all steps) ----------
// All 3 gates per wave (shared gathered A). 2048*64 tiles; 8 waves/block.
// U stored swizzled: U[((g*2048+mt)*64+nt)*256 + lane*8 + e] (bf16).
__global__ __launch_bounds__(256) void k_uproj(const int* __restrict__ X,
                                               const __bf16* __restrict__ embbf,
                                               const __bf16* __restrict__ Wxt,
                                               __bf16* __restrict__ U) {
  const int lane = threadIdx.x & 31;
  const int wid  = blockIdx.x * 8 + (threadIdx.x >> 5);
  const int nt = wid & 63;
  const int mt = wid >> 6;           // 0..2047
  const int row0 = mt * 16;
  const __bf16* B0 = Wxt + (size_t)0 * kDL * kDI + (size_t)(nt * 16) * kDI;
  const __bf16* B1 = Wxt + (size_t)1 * kDL * kDI + (size_t)(nt * 16) * kDI;
  const __bf16* B2 = Wxt + (size_t)2 * kDL * kDI + (size_t)(nt * 16) * kDI;
  v8f a0 = {0.f,0.f,0.f,0.f,0.f,0.f,0.f,0.f}, a1 = a0, a2 = a0;
#pragma unroll 4
  for (int kk = 0; kk < kDI; kk += 32) {
    // locality 3 -> WGP-scope prefetch: pull into all cache levels on miss
    __builtin_prefetch(B0 + (size_t)(lane & 15) * kDI + kk + 64, 0, 3);
    __builtin_prefetch(B1 + (size_t)(lane & 15) * kDI + kk + 64, 0, 3);
    __builtin_prefetch(B2 + (size_t)(lane & 15) * kDI + kk + 64, 0, 3);
    v16bf a  = load_a_gather(embbf, X, row0, kk, lane);
    v16bf b0 = load_b_frag(B0 + kk, kDI, lane);
    v16bf b1 = load_b_frag(B1 + kk, kDI, lane);
    v16bf b2 = load_b_frag(B2 + kk, kDI, lane);
    a0 = wmma_bf16(a, b0, a0);
    a1 = wmma_bf16(a, b1, a1);
    a2 = wmma_bf16(a, b2, a2);
  }
  const size_t tb = (((size_t)mt * 64 + nt) << 8) + (lane << 3);
  v8bf u0, u1, u2;
#pragma unroll
  for (int e = 0; e < 8; ++e) {
    u0[e] = (__bf16)a0[e]; u1[e] = (__bf16)a1[e]; u2[e] = (__bf16)a2[e];
  }
  *(v8bf*)(U + tb)                                = u0;  // 16B store
  *(v8bf*)(U + (size_t)1 * kMTg * kNT * 256 + tb) = u1;
  *(v8bf*)(U + (size_t)2 * kMTg * kNT * 256 + tb) = u2;
}

// ---------------- persistent recurrent kernel ------------------------------
// 32 blocks x 256 threads = 256 waves (one per 16x16 tile in each phase).
__global__ __launch_bounds__(256) void k_recur(const __bf16* __restrict__ Wht,
                                               const __bf16* __restrict__ U,
                                               float* hsw0, float* hsw1,
                                               __bf16* hbf0, __bf16* hbf1,
                                               float* __restrict__ zbuf,
                                               __bf16* __restrict__ rhbf,
                                               unsigned* __restrict__ cnt,
                                               float* __restrict__ Hout) {
  const int lane = threadIdx.x & 31;
  const int wid  = blockIdx.x * 8 + (threadIdx.x >> 5);   // 0..255
  const unsigned NBLK = gridDim.x;
  const int mt = wid >> 6;   // 4 row tiles (B=64)
  const int nt = wid & 63;   // 64 col tiles (DL=1024)
  const int tb = (((mt << 6) + nt) << 8) + (lane << 3);   // swizzled tile base
  const size_t Ustride = (size_t)kMTg * kNT * 256;

  for (int t = 0; t < kSEQ; ++t) {
    const float*  hpsw = (t & 1) ? hsw1 : hsw0;
    float*        hnsw = (t & 1) ? hsw0 : hsw1;
    const __bf16* hb   = (t & 1) ? hbf1 : hbf0;
    __bf16*       hnb  = (t & 1) ? hbf0 : hbf1;
    // swizzled U tile base for this (t, mt, nt): global mtile = t*4 + mt
    const size_t utb = ((((size_t)t * 4 + mt) * 64 + nt) << 8) + (lane << 3);

    // -------- phase 1: z and r fused (shared A fragment) --------
    {
      const __bf16* Bz = Wht + (size_t)(nt * 16) * kDL;
      const __bf16* Br = Wht + (size_t)kDL * kDL + (size_t)(nt * 16) * kDL;
      const __bf16* At = hb + (size_t)(mt * 16) * kDL;
      v8f az = {0.f,0.f,0.f,0.f,0.f,0.f,0.f,0.f}, ar = az;
#pragma unroll 4
      for (int kk = 0; kk < kDL; kk += 32) {
        __builtin_prefetch(Bz + (size_t)(lane & 15) * kDL + kk + 64, 0, 3);
        __builtin_prefetch(Br + (size_t)(lane & 15) * kDL + kk + 64, 0, 3);
        v16bf a  = load_a_frag(At + kk, kDL, lane);
        v16bf bz = load_b_frag(Bz + kk, kDL, lane);
        v16bf br = load_b_frag(Br + kk, kDL, lane);
        az = wmma_bf16(a, bz, az);
        ar = wmma_bf16(a, br, ar);
      }
      v8bf uz = *(const v8bf*)(U + utb);                 // 16B coalesced
      v8bf ur = *(const v8bf*)(U + Ustride + utb);
      v8f  hv = *(const v8f*)(hpsw + tb);                // 32B coalesced
      v8f  zv;
#pragma unroll
      for (int e = 0; e < 8; ++e) {
        float z = 1.f / (1.f + __expf(-(az[e] + (float)uz[e])));
        float r = 1.f / (1.f + __expf(-(ar[e] + (float)ur[e])));
        zv[e] = z;
        int m = mt * 16 + e + ((lane & 16) ? 8 : 0);
        int n = nt * 16 + (lane & 15);
        rhbf[m * kDL + n] = (__bf16)(r * hv[e]);         // row-major for A reload
      }
      *(v8f*)(zbuf + tb) = zv;                           // 32B coalesced
    }
    grid_barrier(&cnt[2 * t], NBLK);

    // -------- phase 2: candidate + state update (dual K accumulators) ------
    {
      const __bf16* Bc = Wht + (size_t)2 * kDL * kDL + (size_t)(nt * 16) * kDL;
      const __bf16* At = rhbf + (size_t)(mt * 16) * kDL;
      v8f ac0 = {0.f,0.f,0.f,0.f,0.f,0.f,0.f,0.f}, ac1 = ac0;
#pragma unroll 4
      for (int kk = 0; kk < kDL; kk += 64) {
        __builtin_prefetch(Bc + (size_t)(lane & 15) * kDL + kk + 128, 0, 3);
        v16bf a0 = load_a_frag(At + kk, kDL, lane);
        v16bf b0 = load_b_frag(Bc + kk, kDL, lane);
        v16bf a1 = load_a_frag(At + kk + 32, kDL, lane);
        v16bf b1 = load_b_frag(Bc + kk + 32, kDL, lane);
        ac0 = wmma_bf16(a0, b0, ac0);
        ac1 = wmma_bf16(a1, b1, ac1);
      }
      v8bf uc = *(const v8bf*)(U + 2 * Ustride + utb);
      v8f  zv = *(const v8f*)(zbuf + tb);
      v8f  hv = *(const v8f*)(hpsw + tb);
      v8f  ov;
#pragma unroll
      for (int e = 0; e < 8; ++e) {
        float cand = tanhf(ac0[e] + ac1[e] + (float)uc[e]);
        float out  = (1.f - zv[e]) * hv[e] + zv[e] * cand;
        ov[e] = out;
        int m = mt * 16 + e + ((lane & 16) ? 8 : 0);
        int n = nt * 16 + (lane & 15);
        hnb[m * kDL + n] = (__bf16)out;                  // row-major for A reload
        Hout[(size_t)t * kB * kDL + (size_t)m * kDL + n] = out;  // canonical out
      }
      *(v8f*)(hnsw + tb) = ov;                           // 32B coalesced
    }
    grid_barrier(&cnt[2 * t + 1], NBLK);
  }
}

// ---------------- host launcher --------------------------------------------
extern "C" void kernel_launch(void* const* d_in, const int* in_sizes, int n_in,
                              void* d_out, int out_size, void* d_ws, size_t ws_size,
                              hipStream_t stream) {
  const int*   X   = (const int*)d_in[0];
  const float* h   = (const float*)d_in[1];
  const float* emb = (const float*)d_in[2];
  const float* Wz  = (const float*)d_in[3];
  const float* Wr  = (const float*)d_in[4];
  const float* Wh  = (const float*)d_in[5];
  float* Hout = (float*)d_out;

  char* ws = (char*)d_ws;
  size_t off = 0;
  auto alloc = [&](size_t bytes) {
    size_t o = off;
    off = (off + bytes + 255) & ~(size_t)255;
    return o;
  };
  unsigned* cnt   = (unsigned*)(ws + alloc((size_t)2 * kSEQ * sizeof(unsigned)));
  __bf16*   embbf = (__bf16*)  (ws + alloc((size_t)kV * kDI * 2));
  __bf16*   Wxt   = (__bf16*)  (ws + alloc((size_t)3 * kDL * kDI * 2));
  __bf16*   Wht   = (__bf16*)  (ws + alloc((size_t)3 * kDL * kDL * 2));
  __bf16*   U     = (__bf16*)  (ws + alloc((size_t)3 * kSEQ * kB * kDL * 2));
  float*    hsw0  = (float*)   (ws + alloc((size_t)kB * kDL * 4));
  float*    hsw1  = (float*)   (ws + alloc((size_t)kB * kDL * 4));
  __bf16*   hbf0  = (__bf16*)  (ws + alloc((size_t)kB * kDL * 2));
  __bf16*   hbf1  = (__bf16*)  (ws + alloc((size_t)kB * kDL * 2));
  float*    zbuf  = (float*)   (ws + alloc((size_t)kB * kDL * 4));
  __bf16*   rhbf  = (__bf16*)  (ws + alloc((size_t)kB * kDL * 2));
  (void)ws_size; (void)n_in; (void)in_sizes; (void)out_size;

  {
    long n = (long)kV * kDI;
    k_emb2bf<<<(unsigned)((n + 255) / 256), 256, 0, stream>>>(emb, embbf, n);
  }
  {
    long n = 3L * kDL * kDI;
    k_wx<<<(unsigned)((n + 255) / 256), 256, 0, stream>>>(Wz, Wr, Wh, Wxt);
  }
  {
    long n = 3L * kDL * kDL;
    k_wh<<<(unsigned)((n + 255) / 256), 256, 0, stream>>>(Wz, Wr, Wh, Wht);
  }
  k_init<<<(kB * kDL + 255) / 256, 256, 0, stream>>>(h, hsw0, hbf0, cnt);

  // 2048*64 tiles, 3 gates per wave, 8 waves per block
  k_uproj<<<(kMTg * kNT) / 8, 256, 0, stream>>>(X, embbf, Wxt, U);

  // persistent recurrent kernel: 32 blocks co-resident, software barriers
  k_recur<<<32, 256, 0, stream>>>(Wht, U, hsw0, hsw1, hbf0, hbf1, zbuf, rhbf,
                                  cnt, Hout);
}
